// LossOnlyWeightGNN_38878043964037
// MI455X (gfx1250) — compile-verified, matrix operand order checked by reading, not run on hardware
//
#include <hip/hip_runtime.h>
#include <hip/hip_bf16.h>

#define DIM 128
#define EMB 8
#define HID 128
#define NNODES 50000
#define NEDGES 400000
#define IN_DIM (3*DIM+EMB)   // 392
#define KPAD 416             // 13 * 32 (K padded with zeros)
#define KT_E 13              // K-tiles for edge MLP layer 1
#define KT_M 4               // K-tiles for 128-deep matmuls
#define NT 8                 // N-tiles (128 / 16)

typedef __attribute__((ext_vector_type(16))) _Float16 v16h;
typedef __attribute__((ext_vector_type(8)))  _Float16 v8h;
typedef __attribute__((ext_vector_type(8)))  float    v8f;

__device__ __forceinline__ v16h cat8(v8h lo, v8h hi) {
  return __builtin_shufflevector(lo, hi, 0,1,2,3,4,5,6,7,8,9,10,11,12,13,14,15);
}

__device__ __forceinline__ v8f zero8() {
  v8f z = {0.f,0.f,0.f,0.f,0.f,0.f,0.f,0.f};
  return z;
}

// ---------------- utility kernels ----------------

__global__ void zero_f(float* __restrict__ p, int n) {
  int i = blockIdx.x * blockDim.x + threadIdx.x;
  if (i < n) p[i] = 0.0f;
}

// Pack a row-major f32 weight [K x 128] into per-lane WMMA B-fragment layout (f16).
// dst[((kt*8+nt)*32 + lane)*16 + i] = W[k(lane,i), nt*16 + (lane&15)], zero-padded in K.
__global__ void pack_w(const float* __restrict__ w, int K, int ktiles,
                       _Float16* __restrict__ dst) {
  int tid = blockIdx.x * blockDim.x + threadIdx.x;
  int total = ktiles * NT * 32 * 16;
  if (tid >= total) return;
  int i  = tid & 15;
  int l  = (tid >> 4) & 31;
  int nt = (tid >> 9) & 7;
  int kt = tid >> 12;
  int kb = (l >> 4) * 8;
  int k  = kt * 32 + ((i < 8) ? (kb + i) : (16 + kb + (i - 8)));
  int n  = nt * 16 + (l & 15);
  dst[tid] = (k < K) ? (_Float16)w[k * HID + n] : (_Float16)0.0f;
}

__global__ void deg_k(const int* __restrict__ ei, float* __restrict__ deg) {
  int e = blockIdx.x * blockDim.x + threadIdx.x;
  if (e < NEDGES) atomicAdd(&deg[ei[e]], 1.0f);
}

__global__ void segmax_k(const int* __restrict__ ei, const float* __restrict__ w,
                         unsigned int* __restrict__ m) {
  int e = blockIdx.x * blockDim.x + threadIdx.x;
  if (e < NEDGES) atomicMax(&m[ei[e]], __float_as_uint(w[e]));  // w>0: uint order == float order
}

__global__ void segsum_k(const int* __restrict__ ei, const float* __restrict__ w,
                         const float* __restrict__ m, float* __restrict__ ssum) {
  int e = blockIdx.x * blockDim.x + threadIdx.x;
  if (e < NEDGES) {
    int s = ei[e];
    atomicAdd(&ssum[s], expf(w[e] - m[s]));
  }
}

__global__ void alpha_k(const int* __restrict__ ei, const float* __restrict__ w,
                        const float* __restrict__ m, const float* __restrict__ ssum,
                        float* __restrict__ alpha) {
  int e = blockIdx.x * blockDim.x + threadIdx.x;
  if (e < NEDGES) {
    int s = ei[e];
    alpha[e] = expf(w[e] - m[s]) / (ssum[s] + 1e-12f);
  }
}

// ---------------- main edge kernel: 1 wave per 16-edge tile ----------------

__launch_bounds__(32)
__global__ void edge_main(const float* __restrict__ z,
                          const int* __restrict__ ei, const int* __restrict__ et,
                          const float* __restrict__ temb,
                          const _Float16* __restrict__ ew1p,
                          const float* __restrict__ eb1, const float* __restrict__ ew2,
                          const float* __restrict__ eb2,
                          const _Float16* __restrict__ mw1p, const float* __restrict__ mb1,
                          const _Float16* __restrict__ mw2p, const float* __restrict__ mb2,
                          const float* __restrict__ deg,
                          float* __restrict__ wbuf, float* __restrict__ agg) {
  __shared__ _Float16 ein[16 * KPAD];   // e_in tile, f16, 16 rows x 416 cols
  __shared__ _Float16 h1[16 * HID];     // msg-MLP hidden tile
  __shared__ int   sS[16];
  __shared__ int   sD[16];
  __shared__ int   sT[16];
  __shared__ float sScale[16];

  const int lane = threadIdx.x;
  const int base = blockIdx.x * 16;

  if (lane < 16) {
    int e = base + lane;
    int s = ei[e];
    sS[lane] = s;
    sD[lane] = ei[NEDGES + e];
    sT[lane] = et[e];
    sScale[lane] = 1.0f / (deg[s] + 1e-12f);
  }
  __syncthreads();

  // ---- stage e_in: phase 1, cols 0..255 = [zi | zj] ----
  for (int idx = lane; idx < 16 * 256; idx += 32) {
    int row = idx >> 8;
    int col = idx & 255;
    float v = (col < DIM) ? z[(long)sS[row] * DIM + col]
                          : z[(long)sD[row] * DIM + (col - DIM)];
    ein[row * KPAD + col] = (_Float16)v;
  }
  __syncthreads();
  // ---- phase 2: cols 256..415 = [|zi-zj| | type_emb | zero pad] (from LDS) ----
  for (int idx = lane; idx < 16 * 160; idx += 32) {
    int row = idx / 160;
    int j   = idx - row * 160;
    _Float16 v;
    if (j < DIM) {
      float a = (float)ein[row * KPAD + j];
      float b = (float)ein[row * KPAD + DIM + j];
      v = (_Float16)fabsf(a - b);
    } else if (j < DIM + EMB) {
      v = (_Float16)temb[sT[row] * EMB + (j - DIM)];
    } else {
      v = (_Float16)0.0f;
    }
    ein[row * KPAD + 256 + j] = v;
  }
  __syncthreads();

  const int half = lane >> 4;   // 0 or 1 (selects K sub-range of A/B fragments)
  const int nl   = lane & 15;   // A row / C column within tile
  const int kb   = half * 8;

  // ---- edge MLP layer 1: [16x416] @ [416x128], 13x8 WMMAs ----
  v8f acc[NT];
  #pragma unroll
  for (int t = 0; t < NT; t++) acc[t] = zero8();

  #pragma unroll
  for (int kt = 0; kt < KT_E; kt++) {
    // batch-load all 8 B fragments first -> one load clause, one wait, 8 WMMAs
    v16h bf[NT];
    #pragma unroll
    for (int nt = 0; nt < NT; nt++)
      bf[nt] = *(const v16h*)&ew1p[((kt * NT + nt) * 32 + lane) * 16];
    v8h lo = *(const v8h*)&ein[nl * KPAD + kt * 32 + kb];
    v8h hi = *(const v8h*)&ein[nl * KPAD + kt * 32 + 16 + kb];
    v16h af = cat8(lo, hi);
    #pragma unroll
    for (int nt = 0; nt < NT; nt++)
      acc[nt] = __builtin_amdgcn_wmma_f32_16x16x32_f16(
          false, af, false, bf[nt], (short)0, acc[nt], false, false);
  }

  // ---- edge MLP layer 2: relu(h+eb1) . ew2, row-wise shuffle reduction ----
  {
    float p[8];
    #pragma unroll
    for (int r = 0; r < 8; r++) p[r] = 0.f;
    #pragma unroll
    for (int nt = 0; nt < NT; nt++) {
      int n = nt * 16 + nl;
      float b1 = eb1[n];
      float w2 = ew2[n];
      #pragma unroll
      for (int r = 0; r < 8; r++) {
        float h = acc[nt][r] + b1;
        h = h > 0.f ? h : 0.f;
        p[r] += h * w2;
      }
    }
    #pragma unroll
    for (int r = 0; r < 8; r++) {
      float v = p[r];
      v += __shfl_xor(v, 1, 16);
      v += __shfl_xor(v, 2, 16);
      v += __shfl_xor(v, 4, 16);
      v += __shfl_xor(v, 8, 16);
      p[r] = v;
    }
    if (nl == 0) {
      float b2 = eb2[0];
      #pragma unroll
      for (int r = 0; r < 8; r++) {
        int m = r + half * 8;
        float x = p[r] + b2;
        wbuf[base + m] = 1.0f / (1.0f + expf(-x));   // sigmoid
      }
    }
  }

  // ---- msg MLP layer 1: zj @ mw1 (A = e_in cols 128..255), 4x8 WMMAs ----
  v8f acc2[NT];
  #pragma unroll
  for (int t = 0; t < NT; t++) acc2[t] = zero8();

  #pragma unroll
  for (int kt = 0; kt < KT_M; kt++) {
    v16h bf[NT];
    #pragma unroll
    for (int nt = 0; nt < NT; nt++)
      bf[nt] = *(const v16h*)&mw1p[((kt * NT + nt) * 32 + lane) * 16];
    v8h lo = *(const v8h*)&ein[nl * KPAD + DIM + kt * 32 + kb];
    v8h hi = *(const v8h*)&ein[nl * KPAD + DIM + kt * 32 + 16 + kb];
    v16h af = cat8(lo, hi);
    #pragma unroll
    for (int nt = 0; nt < NT; nt++)
      acc2[nt] = __builtin_amdgcn_wmma_f32_16x16x32_f16(
          false, af, false, bf[nt], (short)0, acc2[nt], false, false);
  }

  // relu + mb1, spill to LDS as f16 A-source for the second matmul
  #pragma unroll
  for (int nt = 0; nt < NT; nt++) {
    int n = nt * 16 + nl;
    float b1 = mb1[n];
    #pragma unroll
    for (int r = 0; r < 8; r++) {
      float h = acc2[nt][r] + b1;
      h = h > 0.f ? h : 0.f;
      h1[(r + half * 8) * HID + n] = (_Float16)h;
    }
  }
  __syncthreads();

  // ---- msg MLP layer 2: h1 @ mw2, 4x8 WMMAs ----
  v8f acc3[NT];
  #pragma unroll
  for (int t = 0; t < NT; t++) acc3[t] = zero8();

  #pragma unroll
  for (int kt = 0; kt < KT_M; kt++) {
    v16h bf[NT];
    #pragma unroll
    for (int nt = 0; nt < NT; nt++)
      bf[nt] = *(const v16h*)&mw2p[((kt * NT + nt) * 32 + lane) * 16];
    v8h lo = *(const v8h*)&h1[nl * HID + kt * 32 + kb];
    v8h hi = *(const v8h*)&h1[nl * HID + kt * 32 + 16 + kb];
    v16h af = cat8(lo, hi);
    #pragma unroll
    for (int nt = 0; nt < NT; nt++)
      acc3[nt] = __builtin_amdgcn_wmma_f32_16x16x32_f16(
          false, af, false, bf[nt], (short)0, acc3[nt], false, false);
  }

  // ---- scale by 1/deg(src) and scatter-add into agg ----
  #pragma unroll
  for (int r = 0; r < 8; r++) {
    int m = r + half * 8;
    int s = sS[m];
    float sc = sScale[m];
    float* aggrow = agg + (long)s * DIM;
    #pragma unroll
    for (int nt = 0; nt < NT; nt++) {
      int n = nt * 16 + nl;
      float v = (acc3[nt][r] + mb2[n]) * sc;
      atomicAdd(&aggrow[n], v);
    }
  }
}

// ---------------- LayerNorm: 1 wave per node, agg already in `out` ----------------

__launch_bounds__(256)
__global__ void ln_k(const float* __restrict__ z, const float* __restrict__ gamma,
                     const float* __restrict__ beta, float* __restrict__ out) {
  int wid  = threadIdx.x >> 5;
  int lane = threadIdx.x & 31;
  long node = (long)blockIdx.x * 8 + wid;
  if (node >= NNODES) return;
  const float* zr = z + node * DIM;
  float* orow = out + node * DIM;

  float x[4];
  float s = 0.f;
  #pragma unroll
  for (int j = 0; j < 4; j++) {
    int c = lane * 4 + j;
    x[j] = zr[c] + orow[c];      // x = z + agg
    s += x[j];
  }
  #pragma unroll
  for (int off = 1; off < 32; off <<= 1) s += __shfl_xor(s, off, 32);
  float mu = s * (1.0f / DIM);

  float v = 0.f;
  #pragma unroll
  for (int j = 0; j < 4; j++) { float d = x[j] - mu; v += d * d; }
  #pragma unroll
  for (int off = 1; off < 32; off <<= 1) v += __shfl_xor(v, off, 32);
  float inv = rsqrtf(v * (1.0f / DIM) + 1e-5f);

  #pragma unroll
  for (int j = 0; j < 4; j++) {
    int c = lane * 4 + j;
    orow[c] = (x[j] - mu) * inv * gamma[c] + beta[c];
  }
}

// ---------------- launcher ----------------

extern "C" void kernel_launch(void* const* d_in, const int* in_sizes, int n_in,
                              void* d_out, int out_size, void* d_ws, size_t ws_size,
                              hipStream_t stream) {
  const float* z    = (const float*)d_in[0];
  const int*   ei   = (const int*)d_in[1];     // [2, E] : src, dst
  const int*   et   = (const int*)d_in[2];
  const float* temb = (const float*)d_in[3];
  const float* ew1  = (const float*)d_in[4];
  const float* eb1  = (const float*)d_in[5];
  const float* ew2  = (const float*)d_in[6];
  const float* eb2  = (const float*)d_in[7];
  const float* mw1  = (const float*)d_in[8];
  const float* mb1  = (const float*)d_in[9];
  const float* mw2  = (const float*)d_in[10];
  const float* mb2  = (const float*)d_in[11];
  const float* gamma= (const float*)d_in[12];
  const float* beta = (const float*)d_in[13];

  char* ws = (char*)d_ws;
  size_t off = 0;
  auto walloc = [&](size_t bytes) -> void* {
    void* p = ws + off;
    off = (off + bytes + 255) & ~(size_t)255;
    return p;
  };
  float* deg  = (float*)walloc((size_t)NNODES * 4);
  float* maxw = (float*)walloc((size_t)NNODES * 4);
  float* ssum = (float*)walloc((size_t)NNODES * 4);
  float* wbuf = (float*)walloc((size_t)NEDGES * 4);
  _Float16* ew1p = (_Float16*)walloc((size_t)KT_E * NT * 32 * 16 * 2);
  _Float16* mw1p = (_Float16*)walloc((size_t)KT_M * NT * 32 * 16 * 2);
  _Float16* mw2p = (_Float16*)walloc((size_t)KT_M * NT * 32 * 16 * 2);

  float* out   = (float*)d_out;                       // [N*DIM] (used as agg first)
  float* alpha = out + (size_t)NNODES * DIM;          // [E]

  // zero accumulation buffers (re-zeroed every call for graph-replay determinism)
  zero_f<<<(NNODES * DIM + 255) / 256, 256, 0, stream>>>(out, NNODES * DIM);
  zero_f<<<(NNODES + 255) / 256, 256, 0, stream>>>(deg, NNODES);
  zero_f<<<(NNODES + 255) / 256, 256, 0, stream>>>(maxw, NNODES);
  zero_f<<<(NNODES + 255) / 256, 256, 0, stream>>>(ssum, NNODES);

  // pack weights to f16 WMMA B-fragment layout
  pack_w<<<(KT_E * NT * 512 + 255) / 256, 256, 0, stream>>>(ew1, IN_DIM, KT_E, ew1p);
  pack_w<<<(KT_M * NT * 512 + 255) / 256, 256, 0, stream>>>(mw1, DIM, KT_M, mw1p);
  pack_w<<<(KT_M * NT * 512 + 255) / 256, 256, 0, stream>>>(mw2, DIM, KT_M, mw2p);

  // degrees
  deg_k<<<(NEDGES + 255) / 256, 256, 0, stream>>>(ei, deg);

  // main fused edge kernel (both MLP branches, WMMA)
  edge_main<<<NEDGES / 16, 32, 0, stream>>>(z, ei, et, temb, ew1p, eb1, ew2, eb2,
                                            mw1p, mb1, mw2p, mb2, deg, wbuf, out);

  // segment softmax over src
  segmax_k<<<(NEDGES + 255) / 256, 256, 0, stream>>>(ei, wbuf, (unsigned int*)maxw);
  segsum_k<<<(NEDGES + 255) / 256, 256, 0, stream>>>(ei, wbuf, maxw, ssum);
  alpha_k<<<(NEDGES + 255) / 256, 256, 0, stream>>>(ei, wbuf, maxw, ssum, alpha);

  // x = z + agg -> LayerNorm -> out
  ln_k<<<(NNODES + 7) / 8, 256, 0, stream>>>(z, gamma, beta, out);
}
